// Net_49830210568756
// MI455X (gfx1250) — matrix operand order repaired and checked by reading, not compiled, not run on hardware
//
#include <hip/hip_runtime.h>
#include <hip/hip_bf16.h>

// ---------------------------------------------------------------------------
// MI455X / gfx1250: wave32, WMMA 16x16x32 f16 -> f32.
// Model is tiny (fits in L2); encoder/decoder recurrences are run as
// persistent single-WGP kernels with LDS-resident state, WMMA for all GEMMs.
// ---------------------------------------------------------------------------

typedef __attribute__((ext_vector_type(16))) _Float16 v16h;
typedef __attribute__((ext_vector_type(8)))  float    v8f;

#define WMMA_F16(a, b, c) \
  __builtin_amdgcn_wmma_f32_16x16x32_f16(false, (a), false, (b), (short)0, (c), false, false)

// A fragment: 16x32 (MxK) f16. lanes 0-15 -> M=lane, lanes 16-31 -> M=lane-16.
// element i<8: K = hi*8 + i ; i>=8: K = 16 + hi*8 + (i-8).
__device__ __forceinline__ v16h load_a_frag(const _Float16* A, int lda,
                                            int k0, int lane, int Mvalid) {
  const int m = lane & 15, hi = lane >> 4;
  v16h a;
  if (m < Mvalid) {
    const _Float16* p = A + m * lda + k0 + hi * 8;
#pragma unroll
    for (int i = 0; i < 8; ++i) { a[i] = p[i]; a[i + 8] = p[16 + i]; }
  } else {
#pragma unroll
    for (int i = 0; i < 16; ++i) a[i] = (_Float16)0.f;
  }
  return a;
}

// B fragment: 32x16 (KxN) f16 from N-major ("BT") storage BT[n*ldb + k].
// element i<8: K = hi*8 + i ; i>=8: K = 16 + hi*8 + (i-8); N = n0 + (lane&15).
__device__ __forceinline__ v16h load_b_frag(const _Float16* BT, int ldb,
                                            int k0, int n0, int lane) {
  const int nn = n0 + (lane & 15), hi = lane >> 4;
  const _Float16* p = BT + nn * ldb + k0 + hi * 8;
  v16h b;
#pragma unroll
  for (int i = 0; i < 8; ++i) { b[i] = p[i]; b[i + 8] = p[16 + i]; }
  return b;
}

// One wave accumulates a 16x16 tile of A(Mx K) * B(K x N) at column n0.
__device__ __forceinline__ v8f wave_gemm_acc(const _Float16* A, int lda, int Mvalid,
                                             const _Float16* BT, int ldb,
                                             int n0, int K, v8f acc, int lane) {
  for (int k0 = 0; k0 < K; k0 += 32) {
    v16h a = load_a_frag(A, lda, k0, lane, Mvalid);
    v16h b = load_b_frag(BT, ldb, k0, n0, lane);
    acc = WMMA_F16(a, b, acc);
  }
  return acc;
}

__device__ __forceinline__ float sigf(float x) { return 1.f / (1.f + __expf(-x)); }
__device__ __forceinline__ float sgf(float x)  { return tanhf(0.5f * x) * 0.5f + 0.5f; }

// ---------------------------------------------------------------------------
// Prep: f32 -> f16 transposed weight layouts (BT[n*Kpad + k]), zero padded.
// ---------------------------------------------------------------------------
__global__ void __launch_bounds__(256) k_prep_bT(const float* __restrict__ src,
                                                 _Float16* __restrict__ dst,
                                                 int K, int N, int Kpad, int Npad) {
  int i = blockIdx.x * blockDim.x + threadIdx.x;
  if (i >= Npad * Kpad) return;
  int n = i / Kpad, k = i - n * Kpad;
  float v = (n < N && k < K) ? src[k * N + n] : 0.f;
  dst[i] = (_Float16)v;
}

// data (8,400,120) f32 -> (3200 x 128) f16, K padded with zeros.
__global__ void __launch_bounds__(256) k_prep_x0(const float* __restrict__ data,
                                                 _Float16* __restrict__ dst) {
  int i = blockIdx.x * blockDim.x + threadIdx.x;
  if (i >= 3200 * 128) return;
  int r = i >> 7, k = i & 127;
  dst[i] = (k < 120) ? (_Float16)data[r * 120 + k] : (_Float16)0.f;
}

// ---------------------------------------------------------------------------
// GEMM + bias: C(M x N) f32 = A(M x K) f16 * BT^T + bias. One 16x16 tile/wave.
// grid.x = M/16, grid.y = N/128 (8 waves per block).
// ---------------------------------------------------------------------------
__global__ void __launch_bounds__(256) k_gemm_bias(const _Float16* __restrict__ A, int lda,
                                                   const _Float16* __restrict__ BT, int ldb,
                                                   const float* __restrict__ bias,
                                                   float* __restrict__ C, int ldc, int K) {
  const int wave = threadIdx.x >> 5, lane = threadIdx.x & 31;
  const int m0 = blockIdx.x * 16;
  const int n0 = (blockIdx.y * 8 + wave) * 16;
  __builtin_prefetch(BT + n0 * ldb, 0, 1);
  v8f acc = {};
  acc = wave_gemm_acc(A + m0 * lda, lda, 16, BT, ldb, n0, K, acc, lane);
  const int n = n0 + (lane & 15), hi = lane >> 4;
  const float bv = bias[n];
#pragma unroll
  for (int v = 0; v < 8; ++v) {
    int m = m0 + v + hi * 8;
    C[m * ldc + n] = acc[v] + bv;
  }
}

// ---------------------------------------------------------------------------
// LSTM scan: one block per direction (blockIdx.x = dir), 8 waves.
// Per step: z = xz[t] + h @ Whh via WMMA; gates; h,c in LDS.
// Wave w owns strided tiles n0 = w*16 + j*128 so gate partners (cols +256k)
// share the same lane/VGPR slot.
// ---------------------------------------------------------------------------
__global__ void __launch_bounds__(256) k_lstm_scan(const float* __restrict__ xz0,
                                                   const float* __restrict__ xz1,
                                                   const _Float16* __restrict__ whhT0,
                                                   const _Float16* __restrict__ whhT1,
                                                   const int* __restrict__ length,
                                                   _Float16* __restrict__ Xout) {
  __shared__ _Float16 hs[8 * 256];
  __shared__ float    cs[8 * 256];
  const int dir = blockIdx.x;
  const float* xz = dir ? xz1 : xz0;
  const _Float16* whhT = dir ? whhT1 : whhT0;
  const int tid = threadIdx.x, wave = tid >> 5, lane = tid & 31;
  for (int i = tid; i < 8 * 256; i += 256) { hs[i] = (_Float16)0.f; cs[i] = 0.f; }
  __syncthreads();

  const int n = lane & 15, hi = lane >> 4;
  const int hc0 = wave * 16 + n, hc1 = 128 + wave * 16 + n;

  for (int t = 0; t < 400; ++t) {
    v8f zero = {};
    v8f acc[8];
#pragma unroll
    for (int j = 0; j < 8; ++j) acc[j] = zero;
    for (int kk = 0; kk < 256; kk += 32) {
      v16h a = load_a_frag(hs, 256, kk, lane, 8);
#pragma unroll
      for (int j = 0; j < 8; ++j) {
        v16h b = load_b_frag(whhT, 256, kk, wave * 16 + j * 128, lane);
        acc[j] = WMMA_F16(a, b, acc[j]);
      }
    }
    float h0v[8], h1v[8], c0v[8], c1v[8];
    int tts[8];
#pragma unroll
    for (int v = 0; v < 8; ++v) {
      int b = v + hi * 8;
      if (b < 8) {
        int len = length[b];
        int tt = (dir == 0) ? t : ((t < len) ? (len - 1 - t) : t);
        tts[v] = tt;
        const float* zr = xz + (b * 400 + tt) * 1024;
        float zi0 = acc[0][v] + zr[hc0];
        float zf0 = acc[2][v] + zr[256 + hc0];
        float zg0 = acc[4][v] + zr[512 + hc0];
        float zo0 = acc[6][v] + zr[768 + hc0];
        float c0 = sigf(zf0) * cs[b * 256 + hc0] + sigf(zi0) * tanhf(zg0);
        c0v[v] = c0; h0v[v] = sigf(zo0) * tanhf(c0);
        float zi1 = acc[1][v] + zr[hc1];
        float zf1 = acc[3][v] + zr[256 + hc1];
        float zg1 = acc[5][v] + zr[512 + hc1];
        float zo1 = acc[7][v] + zr[768 + hc1];
        float c1 = sigf(zf1) * cs[b * 256 + hc1] + sigf(zi1) * tanhf(zg1);
        c1v[v] = c1; h1v[v] = sigf(zo1) * tanhf(c1);
      }
    }
    __syncthreads();
#pragma unroll
    for (int v = 0; v < 8; ++v) {
      int b = v + hi * 8;
      if (b < 8) {
        hs[b * 256 + hc0] = (_Float16)h0v[v];
        hs[b * 256 + hc1] = (_Float16)h1v[v];
        cs[b * 256 + hc0] = c0v[v];
        cs[b * 256 + hc1] = c1v[v];
        int len = length[b];
        int tt = tts[v];
        _Float16 o0 = (tt < len) ? (_Float16)h0v[v] : (_Float16)0.f;
        _Float16 o1 = (tt < len) ? (_Float16)h1v[v] : (_Float16)0.f;
        Xout[(b * 400 + tt) * 512 + dir * 256 + hc0] = o0;
        Xout[(b * 400 + tt) * 512 + dir * 256 + hc1] = o1;
      }
    }
    __syncthreads();
  }
}

// ---------------------------------------------------------------------------
// Decoder: single persistent workgroup, 200 steps, all state in dynamic LDS.
// ---------------------------------------------------------------------------
__global__ void __launch_bounds__(256) k_decoder(
    const _Float16* __restrict__ hpack,  // (3200 x 512) f16
    const float* __restrict__ atth,      // (3200 x 512) f32
    const int* __restrict__ length,
    const _Float16* attsT, const float* attw,
    const _Float16* syT, const _Float16* gyT, const float* gy_b,
    const _Float16* yyT, const float* yy_b,
    const _Float16* ssT, const _Float16* gsT, const float* gs_b,
    const _Float16* ysT, const float* ys_b,
    const _Float16* snT, const _Float16* nnT, const float* nn_b,
    const _Float16* wnT, const float* wn_b,
    const _Float16* danT, const float* dan_b,
    const _Float16* ndaT, const float* nda_b,
    float* __restrict__ out) {
  extern __shared__ char smem[];
  float* sS  = (float*)smem;           // 8x256 state s
  float* sC  = sS + 8 * 256;           // 8x256 state c
  float* sN  = sC + 8 * 256;           // 8x256 state n
  float* sQ  = sN + 8 * 256;           // 8x512 q = s@atts
  float* sE  = sQ + 8 * 512;           // 8x400 scores / alpha
  float* sLP = sE + 8 * 400;           // 8x64 dialog-act logits / logp
  _Float16* g16  = (_Float16*)(sLP + 8 * 64); // 8x512 g (also reused for v)
  _Float16* y16  = g16 + 8 * 512;      // 8x512 y (padded to 512)
  _Float16* s16  = y16 + 8 * 512;      // 8x256 s as f16 A-operand
  _Float16* u16  = s16 + 8 * 256;      // 8x256 u / n_old as f16 A-operand
  _Float16* da16 = u16 + 8 * 256;      // 8x64  logp_da as f16 A-operand

  const int tid = threadIdx.x, wave = tid >> 5, lane = tid & 31;
  const int n = lane & 15, hi = lane >> 4;

  for (int i = tid; i < 8 * 256; i += 256) {
    sS[i] = 0.f; sC[i] = 0.f; sN[i] = 0.f;
    s16[i] = (_Float16)0.f; u16[i] = (_Float16)0.f;
  }
  for (int i = tid; i < 8 * 512; i += 256) {
    sQ[i] = 0.f; g16[i] = (_Float16)0.f; y16[i] = (_Float16)0.f;
  }
  for (int i = tid; i < 8 * 64; i += 256) { da16[i] = (_Float16)0.f; sLP[i] = 0.f; }
  __syncthreads();

  for (int step = 0; step < 200; ++step) {
    // --- s -> f16 A operand ---
    for (int i = tid; i < 8 * 256; i += 256) s16[i] = (_Float16)sS[i];
    __syncthreads();

    // --- q = s @ atts_w (N=512, K=256) ---
    for (int j = 0; j < 4; ++j) {
      int n0 = (wave * 4 + j) * 16;
      v8f acc = {};
      acc = wave_gemm_acc(s16, 256, 8, attsT, 256, n0, 256, acc, lane);
      int nc = n0 + (lane & 15);
#pragma unroll
      for (int v = 0; v < 8; ++v) { int m = v + hi * 8; if (m < 8) sQ[m * 512 + nc] = acc[v]; }
    }
    __syncthreads();

    // --- attention scores e[b,t] = sum_j tanh(q+atth)*attw ---
    for (int i = tid; i < 8 * 400; i += 256) {
      int b = i / 400, t = i - b * 400;
      const float* ah = atth + (b * 400 + t) * 512;
      const float* qb = sQ + b * 512;
      float e = 0.f;
      for (int j = 0; j < 512; ++j) e += tanhf(qb[j] + ah[j]) * attw[j];
      sE[i] = e;
    }
    __syncthreads();

    // --- masked softmax per batch row ---
    if (tid < 8) {
      int len = length[tid];
      float mx = -1e30f;
      for (int t = 0; t < len; ++t) mx = fmaxf(mx, sE[tid * 400 + t]);
      float sum = 0.f;
      for (int t = 0; t < 400; ++t) {
        float a = (t < len) ? __expf(sE[tid * 400 + t] - mx) : 0.f;
        sE[tid * 400 + t] = a; sum += a;
      }
      float inv = 1.f / sum;
      for (int t = 0; t < 400; ++t) sE[tid * 400 + t] *= inv;
    }
    __syncthreads();

    // --- g = sum_t alpha * hpack ---
    for (int i = tid; i < 8 * 512; i += 256) {
      int b = i >> 9, j = i & 511;
      const float* al = sE + b * 400;
      const _Float16* hp = hpack + (size_t)b * 400 * 512 + j;
      float g = 0.f;
      for (int t = 0; t < 400; ++t) g += al[t] * (float)hp[t * 512];
      g16[i] = (_Float16)g;
    }
    __syncthreads();

    // --- u = tanh(s@sy + g@gy + gy_b)  (N=256) ---
    for (int j = 0; j < 2; ++j) {
      int n0 = (wave * 2 + j) * 16;
      v8f acc = {};
      acc = wave_gemm_acc(s16, 256, 8, syT, 256, n0, 256, acc, lane);
      acc = wave_gemm_acc(g16, 512, 8, gyT, 512, n0, 512, acc, lane);
      int nc = n0 + (lane & 15);
      float bv = gy_b[nc];
#pragma unroll
      for (int v = 0; v < 8; ++v) {
        int m = v + hi * 8;
        if (m < 8) u16[m * 256 + nc] = (_Float16)tanhf(acc[v] + bv);
      }
    }
    __syncthreads();

    // --- y = u @ yy + yy_b  (N=512 padded from 500) ---
    for (int j = 0; j < 4; ++j) {
      int n0 = (wave * 4 + j) * 16;
      v8f acc = {};
      acc = wave_gemm_acc(u16, 256, 8, yyT, 256, n0, 256, acc, lane);
      int nc = n0 + (lane & 15);
      float bv = (nc < 500) ? yy_b[nc] : 0.f;
#pragma unroll
      for (int v = 0; v < 8; ++v) {
        int m = v + hi * 8;
        if (m < 8) y16[m * 512 + nc] = (_Float16)((nc < 500) ? (acc[v] + bv) : 0.f);
      }
    }
    __syncthreads();

    // --- youtput = argmax_y ; also reload n_old into u16 ---
    if (tid < 8) {
      float best = -1e30f; int bi = 0;
      for (int j = 0; j < 500; ++j) {
        float v = (float)y16[tid * 512 + j];
        if (v > best) { best = v; bi = j; }
      }
      out[tid * 200 + step] = (float)bi;
    }
    for (int i = tid; i < 8 * 256; i += 256) u16[i] = (_Float16)sN[i];
    __syncthreads();

    // --- dialog-act logits = n @ nda + nda_b (N=48 padded from 43) ---
    if (wave < 3) {
      int n0 = wave * 16;
      v8f acc = {};
      acc = wave_gemm_acc(u16, 256, 8, ndaT, 256, n0, 256, acc, lane);
      int nc = n0 + (lane & 15);
      float bv = (nc < 43) ? nda_b[nc] : 0.f;
#pragma unroll
      for (int v = 0; v < 8; ++v) {
        int m = v + hi * 8;
        if (m < 8) sLP[m * 64 + nc] = acc[v] + bv;
      }
    }
    __syncthreads();

    // --- logp_da = log_softmax(logits) ; Youtput = argmax ---
    if (tid < 8) {
      float mx = -1e30f; int bi = 0;
      for (int j = 0; j < 43; ++j) {
        float v = sLP[tid * 64 + j];
        if (v > mx) { mx = v; bi = j; }
      }
      float sum = 0.f;
      for (int j = 0; j < 43; ++j) sum += __expf(sLP[tid * 64 + j] - mx);
      float lse = mx + __logf(sum);
      for (int j = 0; j < 64; ++j) {
        float lp = (j < 43) ? (sLP[tid * 64 + j] - lse) : 0.f;
        sLP[tid * 64 + j] = lp;
        da16[tid * 64 + j] = (_Float16)lp;
      }
      out[1600 + tid * 200 + step] = (float)bi;
    }
    __syncthreads();

    // --- z = s@ss + g@gs + y@ys + gs_b + ys_b  (N=1024) ; gate update ---
    {
      v8f zero = {};
      v8f zacc[8];
#pragma unroll
      for (int j = 0; j < 8; ++j) zacc[j] = zero;
      for (int j = 0; j < 8; ++j) {
        int n0 = wave * 16 + j * 128;  // strided: gate partners in same slot
        zacc[j] = wave_gemm_acc(s16, 256, 8, ssT, 256, n0, 256, zacc[j], lane);
        zacc[j] = wave_gemm_acc(g16, 512, 8, gsT, 512, n0, 512, zacc[j], lane);
        zacc[j] = wave_gemm_acc(y16, 512, 8, ysT, 512, n0, 512, zacc[j], lane);
      }
      int hc0 = wave * 16 + n, hc1 = 128 + wave * 16 + n;
#pragma unroll
      for (int v = 0; v < 8; ++v) {
        int b = v + hi * 8;
        if (b < 8) {
          float zi0 = zacc[0][v] + gs_b[hc0] + ys_b[hc0];
          float zf0 = zacc[2][v] + gs_b[256 + hc0] + ys_b[256 + hc0];
          float zg0 = zacc[4][v] + gs_b[512 + hc0] + ys_b[512 + hc0];
          float zo0 = zacc[6][v] + gs_b[768 + hc0] + ys_b[768 + hc0];
          float c0 = sgf(zf0) * sC[b * 256 + hc0] + sgf(zi0) * tanhf(zg0);
          sC[b * 256 + hc0] = c0;
          sS[b * 256 + hc0] = sgf(zo0) * tanhf(c0);
          float zi1 = zacc[1][v] + gs_b[hc1] + ys_b[hc1];
          float zf1 = zacc[3][v] + gs_b[256 + hc1] + ys_b[256 + hc1];
          float zg1 = zacc[5][v] + gs_b[512 + hc1] + ys_b[512 + hc1];
          float zo1 = zacc[7][v] + gs_b[768 + hc1] + ys_b[768 + hc1];
          float c1 = sgf(zf1) * sC[b * 256 + hc1] + sgf(zi1) * tanhf(zg1);
          sC[b * 256 + hc1] = c1;
          sS[b * 256 + hc1] = sgf(zo1) * tanhf(c1);
        }
      }
    }
    __syncthreads();

    // --- v = tanh(s_old@sn + n_old@nn + logp_da@dan + nn_b + dan_b) (reuse g16) ---
    for (int j = 0; j < 2; ++j) {
      int n0 = (wave * 2 + j) * 16;
      v8f acc = {};
      acc = wave_gemm_acc(s16, 256, 8, snT, 256, n0, 256, acc, lane);
      acc = wave_gemm_acc(u16, 256, 8, nnT, 256, n0, 256, acc, lane);
      acc = wave_gemm_acc(da16, 64, 8, danT, 64, n0, 64, acc, lane);
      int nc = n0 + (lane & 15);
      float bv = nn_b[nc] + dan_b[nc];
#pragma unroll
      for (int v = 0; v < 8; ++v) {
        int m = v + hi * 8;
        if (m < 8) g16[m * 256 + nc] = (_Float16)tanhf(acc[v] + bv);
      }
    }
    __syncthreads();

    // --- n_new = v @ wn + wn_b ---
    for (int j = 0; j < 2; ++j) {
      int n0 = (wave * 2 + j) * 16;
      v8f acc = {};
      acc = wave_gemm_acc(g16, 256, 8, wnT, 256, n0, 256, acc, lane);
      int nc = n0 + (lane & 15);
      float bv = wn_b[nc];
#pragma unroll
      for (int v = 0; v < 8; ++v) {
        int m = v + hi * 8;
        if (m < 8) sN[m * 256 + nc] = acc[v] + bv;
      }
    }
    __syncthreads();
  }
}

// ---------------------------------------------------------------------------
// Host orchestration
// ---------------------------------------------------------------------------
extern "C" void kernel_launch(void* const* d_in, const int* in_sizes, int n_in,
                              void* d_out, int out_size, void* d_ws, size_t ws_size,
                              hipStream_t stream) {
  (void)in_sizes; (void)n_in; (void)out_size; (void)ws_size;
  const float* data   = (const float*)d_in[0];
  const int*   length = (const int*)d_in[1];
  const float *Wih[8], *Whh[8], *Lb[8];
  for (int i = 0; i < 8; ++i) {
    Wih[i] = (const float*)d_in[2 + 3 * i];
    Whh[i] = (const float*)d_in[2 + 3 * i + 1];
    Lb[i]  = (const float*)d_in[2 + 3 * i + 2];
  }
  int p = 26;
  const float* attw  = (const float*)d_in[p++];
  const float* atts  = (const float*)d_in[p++];
  const float* atthw = (const float*)d_in[p++];
  const float* atthb = (const float*)d_in[p++];
  const float* syw   = (const float*)d_in[p++];
  const float* gyw   = (const float*)d_in[p++];
  const float* gyb   = (const float*)d_in[p++];
  const float* yyw   = (const float*)d_in[p++];
  const float* yyb   = (const float*)d_in[p++];
  const float* ssw   = (const float*)d_in[p++];
  const float* gsw   = (const float*)d_in[p++];
  const float* gsb   = (const float*)d_in[p++];
  const float* ysw   = (const float*)d_in[p++];
  const float* ysb   = (const float*)d_in[p++];
  const float* snw   = (const float*)d_in[p++];
  const float* nnw   = (const float*)d_in[p++];
  const float* nnb   = (const float*)d_in[p++];
  const float* wnw   = (const float*)d_in[p++];
  const float* wnb   = (const float*)d_in[p++];
  const float* danw  = (const float*)d_in[p++];
  const float* danb  = (const float*)d_in[p++];
  const float* ndaw  = (const float*)d_in[p++];
  const float* ndab  = (const float*)d_in[p++];

  // workspace carve-out (256B aligned slabs)
  char* w = (char*)d_ws;
  auto alloc_h = [&](size_t ne) { _Float16* r = (_Float16*)w; w += ((ne * 2 + 255) / 256) * 256; return r; };
  auto alloc_f = [&](size_t ne) { float* r = (float*)w;       w += ((ne * 4 + 255) / 256) * 256; return r; };

  _Float16* X0  = alloc_h(3200 * 128);
  _Float16* Xa  = alloc_h(3200 * 512);
  _Float16* Xb  = alloc_h(3200 * 512);
  _Float16* hpk = alloc_h(3200 * 512);
  float* xz0    = alloc_f((size_t)3200 * 1024);
  float* xz1    = alloc_f((size_t)3200 * 1024);
  float* atthv  = alloc_f((size_t)3200 * 512);
  _Float16 *wihT[8], *whhT[8];
  for (int i = 0; i < 8; ++i) {
    wihT[i] = alloc_h((size_t)1024 * ((i < 2) ? 128 : 512));
    whhT[i] = alloc_h((size_t)1024 * 256);
  }
  _Float16* attsT = alloc_h(512 * 256);
  _Float16* atthT = alloc_h(512 * 512);
  _Float16* syT   = alloc_h(256 * 256);
  _Float16* gyT   = alloc_h(256 * 512);
  _Float16* yyT   = alloc_h(512 * 256);
  _Float16* ssT   = alloc_h(1024 * 256);
  _Float16* gsT   = alloc_h(1024 * 512);
  _Float16* ysT   = alloc_h(1024 * 512);
  _Float16* snT   = alloc_h(256 * 256);
  _Float16* nnT   = alloc_h(256 * 256);
  _Float16* wnT   = alloc_h(256 * 256);
  _Float16* danT  = alloc_h(256 * 64);
  _Float16* ndaT  = alloc_h(48 * 256);

  auto prep = [&](const float* src, _Float16* dst, int K, int N, int Kp, int Np) {
    int tot = Np * Kp;
    k_prep_bT<<<(tot + 255) / 256, 256, 0, stream>>>(src, dst, K, N, Kp, Np);
  };

  k_prep_x0<<<(3200 * 128 + 255) / 256, 256, 0, stream>>>(data, X0);
  for (int l = 0; l < 4; ++l)
    for (int d = 0; d < 2; ++d) {
      int i = 2 * l + d;
      int din = l ? 512 : 120, dinp = l ? 512 : 128;
      prep(Wih[i], wihT[i], din, 1024, dinp, 1024);
      prep(Whh[i], whhT[i], 256, 1024, 256, 1024);
    }
  prep(atts,  attsT, 256, 512, 256, 512);
  prep(atthw, atthT, 512, 512, 512, 512);
  prep(syw,   syT,   256, 256, 256, 256);
  prep(gyw,   gyT,   512, 256, 512, 256);
  prep(yyw,   yyT,   256, 500, 256, 512);
  prep(ssw,   ssT,   256, 1024, 256, 1024);
  prep(gsw,   gsT,   512, 1024, 512, 1024);
  prep(ysw,   ysT,   500, 1024, 512, 1024);
  prep(snw,   snT,   256, 256, 256, 256);
  prep(nnw,   nnT,   256, 256, 256, 256);
  prep(wnw,   wnT,   256, 256, 256, 256);
  prep(danw,  danT,  43, 256, 64, 256);
  prep(ndaw,  ndaT,  256, 43, 256, 48);

  // Encoder: 4 bidirectional layers
  const _Float16* Xin = X0;
  int dinp = 128;
  for (int l = 0; l < 4; ++l) {
    _Float16* Xout = (l == 3) ? hpk : ((l & 1) ? Xb : Xa);
    dim3 g1(200, 8);
    k_gemm_bias<<<g1, 256, 0, stream>>>(Xin, dinp, wihT[2 * l],     dinp, Lb[2 * l],     xz0, 1024, dinp);
    k_gemm_bias<<<g1, 256, 0, stream>>>(Xin, dinp, wihT[2 * l + 1], dinp, Lb[2 * l + 1], xz1, 1024, dinp);
    k_lstm_scan<<<2, 256, 0, stream>>>(xz0, xz1, whhT[2 * l], whhT[2 * l + 1], length, Xout);
    Xin = Xout;
    dinp = 512;
  }

  // atth = hpack @ atth_w + atth_b (loop-invariant)
  k_gemm_bias<<<dim3(200, 4), 256, 0, stream>>>(hpk, 512, atthT, 512, atthb, atthv, 512, 512);

  // Decoder (persistent, dynamic LDS ~80KB)
  size_t dec_smem = (size_t)(8 * 256 * 3 + 8 * 512 + 8 * 400 + 8 * 64) * 4 +
                    (size_t)(8 * 512 * 2 + 8 * 256 * 2 + 8 * 64) * 2;
  k_decoder<<<1, 256, dec_smem, stream>>>(hpk, atthv, length, attsT, attw,
                                          syT, gyT, gyb, yyT, yyb,
                                          ssT, gsT, gsb, ysT, ysb,
                                          snT, nnT, nnb, wnT, wnb,
                                          danT, danb, ndaT, ndab,
                                          (float*)d_out);
}
